// MoE_50972671869718
// MI455X (gfx1250) — matrix operand order repaired and checked by reading, not compiled, not run on hardware
//
#include <hip/hip_runtime.h>
#include <hip/hip_bf16.h>
#include <math.h>

// ---------------------------------------------------------------------------
// MoE (top-2 of 8 experts, SwiGLU) + shared expert for MI455X (gfx1250).
// Two-pass grouped GEMM design:
//   R  : fp32 router -> per-expert token lists (entry = 2*token+slot)
//   G1 : act[entry, I] = silu(x@Wg)*(x@Wu)   (bf16 out, x K-resident in LDS,
//        double-buffered weight panels: 1 barrier / 64-K block)
//   D  : out[token,H] += w * act@Wd          (double-buffered; act tiles via
//        CDNA5 global_load_async_to_lds_b128, overlapped across chunks)
// Shared expert reuses both kernels (dense instantiation) + the same 32MB act
// buffer after the expert down-pass drained (stream order => deterministic).
// All GEMMs: v_wmma_f32_16x16x32_bf16; B operands column-major in LDS so every
// fragment load is 2x ds_load_b128.
// ---------------------------------------------------------------------------

#define T_TOK 4096
#define HDIM  1024
#define NEXP  8
#define IDIM  2048
#define ISDIM 4096

#define XS_STR  1032   // 64 x (1024+8) bf16: rows land on distinct banks, 16B aligned
#define CM_STR  72     // column-major panels: [n][64+8] bf16
#define ACT_STR 72

typedef __attribute__((ext_vector_type(16))) __bf16 v16bf;
typedef __attribute__((ext_vector_type(8)))  float  v8f;

union FragA { v16bf v; __bf16 e[16]; };
union AccF  { v8f   v; float  e[8];  };

// ---- WMMA helpers (CDNA5 16x16x32 bf16 layouts, ISA 7.12.2) ---------------

__device__ __forceinline__ v16bf load_a_frag(const __bf16* p, int stride, int lane) {
  // A (MxK=16x32), row-major LDS: two contiguous 8-half runs -> 2x ds_load_b128.
  FragA f;
  const __bf16* q = p + (lane & 15) * stride + ((lane & 16) ? 8 : 0);
#pragma unroll
  for (int j = 0; j < 16; ++j) f.e[j] = q[(j < 8) ? j : (j + 8)];
  return f.v;
}

__device__ __forceinline__ v16bf load_b_frag_cm(const __bf16* p, int stride, int lane) {
  // B (KxN=32x16), stored column-major: two contiguous 8-half runs -> 2x ds_load_b128.
  FragA f;
  const __bf16* q = p + (lane & 15) * stride + ((lane & 16) ? 8 : 0);
#pragma unroll
  for (int j = 0; j < 16; ++j) f.e[j] = q[(j < 8) ? j : (j + 8)];
  return f.v;
}

__device__ __forceinline__ v8f wmma_bf16(v16bf a, v16bf b, v8f c) {
  return __builtin_amdgcn_wmma_f32_16x16x32_bf16(false, a, false, b, (short)0, c,
                                                 false, false);
}

// ---- packing helpers ------------------------------------------------------

__device__ __forceinline__ unsigned pack2(float a, float b) {
#if __has_builtin(__builtin_amdgcn_cvt_pk_bf16_f32)
  typedef __attribute__((ext_vector_type(2))) __bf16 v2bf;
  union { v2bf v; unsigned u; } z;
  z.v = __builtin_amdgcn_cvt_pk_bf16_f32(a, b);
  return z.u;
#else
  union { __bf16 h[2]; unsigned u; } z;
  z.h[0] = (__bf16)a; z.h[1] = (__bf16)b;
  return z.u;
#endif
}

__device__ __forceinline__ void store8(__bf16* dst, float4 a, float4 b) {
  union { unsigned u[4]; uint4 q; } z;
  z.u[0] = pack2(a.x, a.y); z.u[1] = pack2(a.z, a.w);
  z.u[2] = pack2(b.x, b.y); z.u[3] = pack2(b.z, b.w);
  *(uint4*)dst = z.q;
}

// fast silu: one v_exp + v_rcp instead of the IEEE division expansion
__device__ __forceinline__ float silu_fast(float g) {
  return g * __builtin_amdgcn_rcpf(1.0f + __expf(-g));
}

// ---- CDNA5 async global->LDS copy (bf16 data, no VGPR round-trip) ---------

__device__ __forceinline__ void async_copy_b128(const void* gbase,
                                                unsigned gofs_bytes,
                                                unsigned lds_addr) {
  asm volatile("global_load_async_to_lds_b128 %0, %1, %2 offset:0"
               :: "v"(lds_addr), "v"(gofs_bytes),
                  "s"((unsigned long long)gbase)
               : "memory");
}
__device__ __forceinline__ void wait_asynccnt0() {
  asm volatile("s_wait_asynccnt 0x0" ::: "memory");
}

// ---------------------------------------------------------------------------
// Router: fp32 logits, top-2, softmax-over-selected. One wave per token.
// ---------------------------------------------------------------------------
__global__ void moe_router_kernel(const float* __restrict__ x,
                                  const float* __restrict__ rw,
                                  int* __restrict__ counts,
                                  int* __restrict__ lists,
                                  float* __restrict__ lweights) {
  const int lane = threadIdx.x & 31;
  const int wave = threadIdx.x >> 5;
  const int t    = blockIdx.x * 8 + wave;

  float acc[NEXP];
#pragma unroll
  for (int e = 0; e < NEXP; ++e) acc[e] = 0.0f;
  for (int h = lane; h < HDIM; h += 32) {
    const float xv = x[(size_t)t * HDIM + h];
#pragma unroll
    for (int e = 0; e < NEXP; ++e) acc[e] += xv * rw[e * HDIM + h];
  }
#pragma unroll
  for (int e = 0; e < NEXP; ++e)
#pragma unroll
    for (int off = 16; off > 0; off >>= 1)
      acc[e] += __shfl_xor(acc[e], off, 32);

  if (lane == 0) {
    int e0 = 0; float v0 = acc[0];
#pragma unroll
    for (int e = 1; e < NEXP; ++e) if (acc[e] > v0) { v0 = acc[e]; e0 = e; }
    int e1 = -1; float v1 = -3.4e38f;
#pragma unroll
    for (int e = 0; e < NEXP; ++e)
      if (e != e0 && acc[e] > v1) { v1 = acc[e]; e1 = e; }
    const float w0 = 1.0f / (1.0f + __expf(v1 - v0));
    const float w1 = 1.0f - w0;
    int p0 = atomicAdd(&counts[e0], 1);
    lists[e0 * T_TOK + p0] = 2 * t + 0;  lweights[e0 * T_TOK + p0] = w0;
    int p1 = atomicAdd(&counts[e1], 1);
    lists[e1 * T_TOK + p1] = 2 * t + 1;  lweights[e1 * T_TOK + p1] = w1;
  }
}

// ---------------------------------------------------------------------------
// Pass 1: act[entry, 0..istride) = silu(x@Wg) * (x@Wu), bf16 output.
// x tile K-resident (129KB LDS) => each weight element read from global once
// per block. Weight panels double-buffered: one barrier per 64-K block, global
// loads + cvt_pk overlap the WMMA stream of the previous panel.
// ---------------------------------------------------------------------------
template <bool GROUPED>
__global__ __launch_bounds__(256, 1)
void moe_gateup_kernel(const float* __restrict__ x,
                       const float* __restrict__ Wg_all,
                       const float* __restrict__ Wu_all,
                       const int* __restrict__ lists,
                       const int* __restrict__ counts,
                       __bf16* __restrict__ actbuf,
                       int istride) {
  __shared__ __bf16 xs[64 * XS_STR];        // [row][K] row-major (A side)
  __shared__ __bf16 wgs[2][64 * CM_STR];    // [i][k] column-major, ping-pong
  __shared__ __bf16 wus[2][64 * CM_STR];
  __shared__ __bf16 acts[64 * ACT_STR];     // staging for coalesced store
  __shared__ int    toks[64];

  const int tid  = threadIdx.x;
  const int lane = tid & 31;
  const int wave = tid >> 5;
  const int tb   = blockIdx.x;

  const float* Wg = Wg_all;
  const float* Wu = Wu_all;
  int cnt = gridDim.x * 64;
  if (GROUPED) {
    const int e = blockIdx.z;
    cnt = counts[e];
    if (tb * 64 >= cnt) return;
    Wg = Wg_all + (size_t)e * HDIM * istride;
    Wu = Wu_all + (size_t)e * HDIM * istride;
  }
  if (tid < 64) {
    const int gi = tb * 64 + tid;
    toks[tid] = GROUPED ? ((gi < cnt) ? lists[blockIdx.z * T_TOK + gi] : -1) : gi;
  }
  __syncthreads();

  // Stage full x tile [64][1024] fp32 -> bf16 (vectorized, gathered rows).
#pragma unroll 4
  for (int s = 0; s < 32; ++s) {
    const int id8 = s * 256 + tid;
    const int r = id8 >> 7, c = (id8 & 127) * 8;
    const int entry = toks[r];
    if (entry >= 0) {
      const int tok = GROUPED ? (entry >> 1) : entry;
      const float4* p = (const float4*)(x + (size_t)tok * HDIM + c);
      store8(&xs[r * XS_STR + c], p[0], p[1]);
    } else {
      *(uint4*)&xs[r * XS_STR + c] = make_uint4(0, 0, 0, 0);
    }
  }

  const int mtg = wave >> 1;          // M-tile (0..3)
  const int ntp = (wave & 1) * 2;     // N-tile pair base (0 or 2)
  const int nch = istride / 64;

#pragma unroll 1
  for (int ic = 0; ic < nch; ++ic) {
    const int icb = ic * 64;

    // Stage one 64-K weight panel (column-major, 4i x 2k micro-tiles).
    auto stage_w = [&](int kbb, int buf) {
#pragma unroll
      for (int s = 0; s < 2; ++s) {
        const int id = s * 256 + tid;
        const int i0 = (id & 15) * 4;
        const int k0 = (id >> 4) * 2;
        const size_t g0 = (size_t)(kbb + k0) * istride + icb + i0;
        const float4 ga = *(const float4*)(Wg + g0);
        const float4 gb = *(const float4*)(Wg + g0 + istride);
        const float gav[4] = {ga.x, ga.y, ga.z, ga.w};
        const float gbv[4] = {gb.x, gb.y, gb.z, gb.w};
        const float4 ua = *(const float4*)(Wu + g0);
        const float4 ub = *(const float4*)(Wu + g0 + istride);
        const float uav[4] = {ua.x, ua.y, ua.z, ua.w};
        const float ubv[4] = {ub.x, ub.y, ub.z, ub.w};
#pragma unroll
        for (int q = 0; q < 4; ++q) {
          *(unsigned*)&wgs[buf][(i0 + q) * CM_STR + k0] = pack2(gav[q], gbv[q]);
          *(unsigned*)&wus[buf][(i0 + q) * CM_STR + k0] = pack2(uav[q], ubv[q]);
        }
      }
    };

    v8f ag[2], au[2];
#pragma unroll
    for (int p = 0; p < 2; ++p) { ag[p] = (v8f)(0.0f); au[p] = (v8f)(0.0f); }

    stage_w(0, 0);
#pragma unroll 1
    for (int kb = 0; kb < HDIM / 64; ++kb) {
      __syncthreads();                      // panel kb visible; kb-1 compute done
      if (kb + 1 < HDIM / 64) stage_w((kb + 1) * 64, (kb + 1) & 1);
      const int bsel = kb & 1;
#pragma unroll
      for (int ks = 0; ks < 2; ++ks) {
        const v16bf A = load_a_frag(&xs[(mtg * 16) * XS_STR + kb * 64 + ks * 32],
                                    XS_STR, lane);
#pragma unroll
        for (int p = 0; p < 2; ++p) {
          const int nt = ntp + p;
          const v16bf Bg = load_b_frag_cm(&wgs[bsel][(nt * 16) * CM_STR + ks * 32],
                                          CM_STR, lane);
          const v16bf Bu = load_b_frag_cm(&wus[bsel][(nt * 16) * CM_STR + ks * 32],
                                          CM_STR, lane);
          ag[p] = wmma_bf16(A, Bg, ag[p]);
          au[p] = wmma_bf16(A, Bu, au[p]);
        }
      }
    }
    __syncthreads();                        // last panel compute done everywhere

    // silu(g)*u -> LDS (u16 scatter, cheap), then coalesced b128 global store.
#pragma unroll
    for (int p = 0; p < 2; ++p) {
      AccF g, u; g.v = ag[p]; u.v = au[p];
      const int rowofs = mtg * 16 + ((lane & 16) ? 8 : 0);
      const int col    = (ntp + p) * 16 + (lane & 15);
#pragma unroll
      for (int r = 0; r < 8; ++r)
        acts[(rowofs + r) * ACT_STR + col] = (__bf16)(silu_fast(g.e[r]) * u.e[r]);
    }
    __syncthreads();
#pragma unroll
    for (int s = 0; s < 2; ++s) {
      const int id8 = s * 256 + tid;
      const int r = id8 >> 3, c = (id8 & 7) * 8;
      const int entry = toks[r];
      if (entry >= 0)
        *(uint4*)(actbuf + (size_t)entry * istride + icb + c) =
            *(const uint4*)&acts[r * ACT_STR + c];
    }
  }
}

// ---------------------------------------------------------------------------
// Pass 2: out[token, 256 H-cols] (+)= w * act[entry,:] @ Wd.
// Double-buffered chunks: act tiles arrive via async global->LDS issued one
// chunk ahead (s_wait_asynccnt at the next loop head), Wd panel conversion
// overlaps the WMMA stream.
// ---------------------------------------------------------------------------
template <bool GROUPED>
__global__ __launch_bounds__(256, 1)
void moe_down_kernel(const __bf16* __restrict__ actbuf,
                     const float* __restrict__ Wd_all,
                     const int* __restrict__ lists,
                     const float* __restrict__ lweights,
                     const int* __restrict__ counts,
                     float* __restrict__ out,
                     int istride) {
  __shared__ __bf16 acts[2][64 * ACT_STR];  // [row][k] row-major, ping-pong
  __shared__ __bf16 wds[2][256 * CM_STR];   // [h][k] column-major, ping-pong
  __shared__ int    toks[64];
  __shared__ float  wsx[64];

  const int tid  = threadIdx.x;
  const int lane = tid & 31;
  const int wave = tid >> 5;
  const int tb   = blockIdx.x;
  const int hgb  = blockIdx.y * 256;

  const float* Wd = Wd_all;
  int cnt = gridDim.x * 64;
  if (GROUPED) {
    const int e = blockIdx.z;
    cnt = counts[e];
    if (tb * 64 >= cnt) return;
    Wd = Wd_all + (size_t)e * istride * HDIM;
  }
  if (tid < 64) {
    const int gi = tb * 64 + tid;
    if (GROUPED) {
      if (gi < cnt) {
        toks[tid] = lists[blockIdx.z * T_TOK + gi];
        wsx[tid]  = lweights[blockIdx.z * T_TOK + gi];
      } else { toks[tid] = -1; wsx[tid] = 0.0f; }
    } else { toks[tid] = gi; wsx[tid] = 1.0f; }
  }
  __syncthreads();

  auto stage_d = [&](int icb, int buf) {
    // act rows (already bf16): async straight into LDS, 16B per lane-op.
#pragma unroll
    for (int s = 0; s < 2; ++s) {
      const int id8 = s * 256 + tid;
      const int r = id8 >> 3, c = (id8 & 7) * 8;
      const int entry = toks[r];
      const unsigned lds = (unsigned)(size_t)&acts[buf][r * ACT_STR + c];
      if (entry >= 0) {
        const unsigned gofs =
            (unsigned)(((size_t)entry * istride + icb + c) * sizeof(__bf16));
        async_copy_b128(actbuf, gofs, lds);
      } else {
        *(uint4*)&acts[buf][r * ACT_STR + c] = make_uint4(0, 0, 0, 0);
      }
    }
    // Wd panel [64 K][256 H] fp32 -> bf16, column-major.
#pragma unroll
    for (int s = 0; s < 8; ++s) {
      const int id = s * 256 + tid;
      const int h0 = (id & 63) * 4;
      const int k0 = (id >> 6) * 2;
      const size_t g0 = (size_t)(icb + k0) * HDIM + hgb + h0;
      const float4 a = *(const float4*)(Wd + g0);
      const float4 b = *(const float4*)(Wd + g0 + HDIM);
      const float av[4] = {a.x, a.y, a.z, a.w};
      const float bv[4] = {b.x, b.y, b.z, b.w};
#pragma unroll
      for (int q = 0; q < 4; ++q)
        *(unsigned*)&wds[buf][(h0 + q) * CM_STR + k0] = pack2(av[q], bv[q]);
    }
  };

  v8f acc[4][2];
#pragma unroll
  for (int m = 0; m < 4; ++m)
#pragma unroll
    for (int p = 0; p < 2; ++p) acc[m][p] = (v8f)(0.0f);

  const int nch = istride / 64;
  stage_d(0, 0);
#pragma unroll 1
  for (int ic = 0; ic < nch; ++ic) {
    wait_asynccnt0();                 // asyncs targeting buffer ic&1 landed
    __syncthreads();                  // ...visible to all waves; ic-1 compute done
    if (ic + 1 < nch) stage_d((ic + 1) * 64, (ic + 1) & 1);
    const int bsel = ic & 1;
#pragma unroll
    for (int ks = 0; ks < 2; ++ks) {
      const int kk = ks * 32;
      v16bf A[4];
#pragma unroll
      for (int m = 0; m < 4; ++m)
        A[m] = load_a_frag(&acts[bsel][(m * 16) * ACT_STR + kk], ACT_STR, lane);
#pragma unroll
      for (int p = 0; p < 2; ++p) {
        const int nt = wave * 2 + p;
        const v16bf B = load_b_frag_cm(&wds[bsel][(nt * 16) * CM_STR + kk],
                                       CM_STR, lane);
#pragma unroll
        for (int m = 0; m < 4; ++m)
          acc[m][p] = wmma_bf16(A[m], B, acc[m][p]);
      }
    }
  }

  // Epilogue. GROUPED: exactly two commutative fp32 atomics per out element
  // (deterministic); dense shared pass runs afterwards -> plain read-add-store.
#pragma unroll
  for (int m = 0; m < 4; ++m)
#pragma unroll
    for (int p = 0; p < 2; ++p) {
      AccF a; a.v = acc[m][p];
      const int rowofs = m * 16 + ((lane & 16) ? 8 : 0);
      const int col    = hgb + (wave * 2 + p) * 16 + (lane & 15);
#pragma unroll
      for (int r = 0; r < 8; ++r) {
        const int row = rowofs + r;
        const int entry = toks[row];
        if (GROUPED) {
          if (entry >= 0)
            atomicAdd(&out[(size_t)(entry >> 1) * HDIM + col],
                      a.e[r] * wsx[row]);
        } else {
          out[(size_t)(tb * 64 + row) * HDIM + col] += a.e[r];
        }
      }
    }
}

// ---------------------------------------------------------------------------
extern "C" void kernel_launch(void* const* d_in, const int* in_sizes, int n_in,
                              void* d_out, int out_size, void* d_ws, size_t ws_size,
                              hipStream_t stream) {
  const float* x  = (const float*)d_in[0];
  const float* rw = (const float*)d_in[1];
  const float* Wg = (const float*)d_in[2];
  const float* Wu = (const float*)d_in[3];
  const float* Wd = (const float*)d_in[4];
  const float* Sg = (const float*)d_in[5];
  const float* Su = (const float*)d_in[6];
  const float* Sd = (const float*)d_in[7];
  float* out = (float*)d_out;

  char*   ws     = (char*)d_ws;
  int*    counts = (int*)ws;
  int*    lists  = (int*)(ws + 256);
  float*  lwts   = (float*)(ws + 256 + (size_t)NEXP * T_TOK * 4);
  __bf16* actbuf = (__bf16*)(ws + (512u << 10));   // 32 MB act buffer (reused)

  hipMemsetAsync(d_out, 0, (size_t)out_size * sizeof(float), stream);
  hipMemsetAsync(counts, 0, NEXP * sizeof(int), stream);

  moe_router_kernel<<<T_TOK / 8, 256, 0, stream>>>(x, rw, counts, lists, lwts);

  // Routed experts (grouped): act = swiglu(x, Wg, Wu); out += w * act@Wd.
  moe_gateup_kernel<true><<<dim3(T_TOK / 64, 1, NEXP), 256, 0, stream>>>(
      x, Wg, Wu, lists, counts, actbuf, IDIM);
  moe_down_kernel<true><<<dim3(T_TOK / 64, HDIM / 256, NEXP), 256, 0, stream>>>(
      actbuf, Wd, lists, lwts, counts, out, IDIM);

  // Shared expert (dense), reusing actbuf after the expert down-pass drained.
  moe_gateup_kernel<false><<<dim3(T_TOK / 64, 1, 1), 256, 0, stream>>>(
      x, Sg, Su, nullptr, nullptr, actbuf, ISDIM);
  moe_down_kernel<false><<<dim3(T_TOK / 64, HDIM / 256, 1), 256, 0, stream>>>(
      actbuf, Sd, nullptr, nullptr, nullptr, out, ISDIM);
}
